// MLP_Channel_38551626449163
// MI455X (gfx1250) — matrix operand order repaired and checked by reading, not compiled, not run on hardware
//
#include <hip/hip_runtime.h>
#include <hip/hip_bf16.h>
#include <math.h>

// ---------------------------------------------------------------------------
// MLP_Channel attention+mixer forward for MI455X (gfx1250, wave32, WMMA)
//   B=64, HEADS=10, CH=300, DIM=3000, H=W=7, S=49
// Workspace layout (floats):
//   qkv0 : [0,            28,224,000)   (B*9000*49)  qkv after 1x1 conv
//   qkv1 : [28,224,000,   56,448,000)   qkv after depthwise conv (q,k,v)
//   wbf  : aliases qkv1 region, bf16 [9216][3008]  (27,721,728 bf16)
//   xbf  : after wbf,        bf16 [3136][3008]     ( 9,433,088 bf16)
//          (both dead before dwconv overwrites qkv1)
//   t    : [0,             9,408,000)   (b,g,s,c)  aliases qkv0
//   y    : [9,408,000,    18,816,000)   LN scratch
// Total ws requirement: 56,448,000 floats = ~226 MB.
// ---------------------------------------------------------------------------

typedef __attribute__((ext_vector_type(16))) __bf16 v16bf;
typedef __attribute__((ext_vector_type(8)))  __bf16 v8bf;
typedef __attribute__((ext_vector_type(8)))  float  v8f;
typedef __attribute__((ext_vector_type(2)))  float  v2f;

#define BATCH   64
#define HEADS   10
#define CH      300
#define DIMK    3000
#define KPAD    3008          // 3000 padded to mult of 32
#define ODIM    9000
#define OPAD    9216          // 9000 padded to mult of 128
#define SPAT    49
#define NTOT    (BATCH*SPAT)  // 3136 = 49*64
#define NROWS   (BATCH*HEADS*SPAT)  // 31360
#define NBGC    (BATCH*HEADS*CH)    // 192000

#if defined(__gfx1250__) && __has_builtin(__builtin_amdgcn_wmma_f32_16x16x4_f32)
#define USE_WMMA_F32 1
#else
#define USE_WMMA_F32 0
#endif

// CDNA5 async global->LDS copy path (ASYNCcnt). Builtin signature (from the
// toolchain diagnostic): (v4i as1*, v4i as3*, imm int offset, imm int cpol).
#if defined(__gfx1250__) && \
    __has_builtin(__builtin_amdgcn_global_load_async_to_lds_b128) && \
    __has_builtin(__builtin_amdgcn_s_wait_asynccnt)
#define USE_ASYNC_COPY 1
typedef int v4i __attribute__((vector_size(16)));
typedef __attribute__((address_space(1))) v4i* gas1_t;
typedef __attribute__((address_space(3))) v4i* las3_t;
#else
#define USE_ASYNC_COPY 0
#endif

__device__ __forceinline__ float gelu_exact(float x) {
  return 0.5f * x * (1.f + erff(x * 0.70710678118654752f));
}

// ---------------------------------------------------------------------------
// 0a) convert w_qkv (f32 [9000][3000]) -> wbf (bf16 [9216][3008], zero padded)
// ---------------------------------------------------------------------------
__global__ __launch_bounds__(256) void conv_w_bf16(
    const float* __restrict__ wq, __bf16* __restrict__ wbf)
{
  int idx = blockIdx.x * 256 + threadIdx.x;     // OPAD * (KPAD/8) = 3,465,216
  if (idx >= OPAD * (KPAD / 8)) return;
  int o  = idx / (KPAD / 8);
  int kc = (idx % (KPAD / 8)) * 8;
  v8bf v;
#pragma unroll
  for (int j = 0; j < 8; ++j) {
    int k = kc + j;
    v[j] = (o < ODIM && k < DIMK) ? (__bf16)wq[(size_t)o * DIMK + k] : (__bf16)0.f;
  }
  *(v8bf*)(wbf + (size_t)o * KPAD + kc) = v;
}

// ---------------------------------------------------------------------------
// 0b) convert+transpose x (f32 [b][c][s]) -> xbf (bf16 [n=b*49+s][KPAD])
// ---------------------------------------------------------------------------
__global__ __launch_bounds__(256) void conv_x_bf16(
    const float* __restrict__ x, __bf16* __restrict__ xbf)
{
  int idx = blockIdx.x * 256 + threadIdx.x;     // NTOT * (KPAD/8) = 1,179,136
  if (idx >= NTOT * (KPAD / 8)) return;
  int n  = idx / (KPAD / 8);
  int kc = (idx % (KPAD / 8)) * 8;
  int b = n / SPAT, s = n % SPAT;
  const float* xp = x + (size_t)b * (DIMK * SPAT) + s;
  v8bf v;
#pragma unroll
  for (int j = 0; j < 8; ++j) {
    int k = kc + j;
    v[j] = (k < DIMK) ? (__bf16)xp[(size_t)k * SPAT] : (__bf16)0.f;
  }
  *(v8bf*)(xbf + (size_t)n * KPAD + kc) = v;
}

// ---------------------------------------------------------------------------
// 1) qkv GEMM: out[b,o,s] = sum_c w[o,c]*x[c,(b,s)] + bias[o]
//    M=9216(o,padded) x N=3136 x K=3008, bf16 WMMA f32 accum.
//    Block: 8 waves x 16 o-rows = 128 M, 64 N; x-panel double-buffered in LDS
//    via async global->LDS copies (ASYNCcnt) when available.
// ---------------------------------------------------------------------------
__global__ __launch_bounds__(256) void qkv_gemm(
    const __bf16* __restrict__ wbf, const __bf16* __restrict__ xbf,
    const float* __restrict__ bq, float* __restrict__ out)
{
  __shared__ __align__(64) __bf16 xls[2][64][32];   // 8 KB double buffer

  const int tid  = threadIdx.x;
  const int lane = tid & 31;
  const int wave = tid >> 5;
  const int mm = lane & 15;
  const int gg = lane >> 4;
  const int n0 = blockIdx.x * 64;
  const int o_base = blockIdx.y * 128 + wave * 16;

  // fill thread mapping: n = tid/4 (0..63), kchunk = (tid&3)*8  (16B each)
  const int fn = tid >> 2;
  const int fk = (tid & 3) << 3;
  const __bf16* fsrc = xbf + (size_t)(n0 + fn) * KPAD + fk;

  const __bf16* wr = wbf + (size_t)(o_base + mm) * KPAD;

  v8f acc0 = {0.f,0.f,0.f,0.f,0.f,0.f,0.f,0.f};
  v8f acc1 = acc0, acc2 = acc0, acc3 = acc0;

  // prologue: fill buffer 0 with kk=0
#if USE_ASYNC_COPY
  __builtin_amdgcn_global_load_async_to_lds_b128(
      (gas1_t)(__bf16*)fsrc, (las3_t)&xls[0][fn][fk], 0, 0);
  __builtin_amdgcn_s_wait_asynccnt(0);
#else
  *(v8bf*)&xls[0][fn][fk] = *(const v8bf*)fsrc;
#endif
  __syncthreads();

  int cur = 0;
  for (int kk = 0; kk < KPAD; kk += 32) {
    if (kk + 32 < KPAD) {   // prefetch next panel into other buffer
#if USE_ASYNC_COPY
      __builtin_amdgcn_global_load_async_to_lds_b128(
          (gas1_t)(__bf16*)(fsrc + kk + 32), (las3_t)&xls[cur ^ 1][fn][fk], 0, 0);
#else
      *(v8bf*)&xls[cur ^ 1][fn][fk] = *(const v8bf*)(fsrc + kk + 32);
#endif
    }
    __builtin_prefetch(wr + kk + 512, 0, 0);   // stream A ahead (global_prefetch_b8)
    // A tile: rows o_base+mm, K = kk + [8g..8g+7] ++ [16+8g..23+8g]
    v8bf a0 = *(const v8bf*)(wr + kk + 8 * gg);
    v8bf a1 = *(const v8bf*)(wr + kk + 16 + 8 * gg);
    v16bf A = __builtin_shufflevector(a0, a1, 0,1,2,3,4,5,6,7,8,9,10,11,12,13,14,15);
#pragma unroll
    for (int nt = 0; nt < 4; ++nt) {
      // B tile: col n0 + nt*16 + mm, K = kk + [8g..8g+7] ++ [16+8g..23+8g]
      v8bf b0 = *(const v8bf*)&xls[cur][nt * 16 + mm][8 * gg];
      v8bf b1 = *(const v8bf*)&xls[cur][nt * 16 + mm][16 + 8 * gg];
      v16bf B = __builtin_shufflevector(b0, b1, 0,1,2,3,4,5,6,7,8,9,10,11,12,13,14,15);
      v8f* accp = nt == 0 ? &acc0 : nt == 1 ? &acc1 : nt == 2 ? &acc2 : &acc3;
      *accp = __builtin_amdgcn_wmma_f32_16x16x32_bf16(false, A, false, B,
                                                      (short)0, *accp, false, false);
    }
#if USE_ASYNC_COPY
    __builtin_amdgcn_s_wait_asynccnt(0);
#endif
    __syncthreads();
    cur ^= 1;
  }

#pragma unroll
  for (int nt = 0; nt < 4; ++nt) {
    const v8f& acc = nt == 0 ? acc0 : nt == 1 ? acc1 : nt == 2 ? acc2 : acc3;
    int n = n0 + nt * 16 + mm;          // < 3136 always
    int bb = n / SPAT, ss = n % SPAT;
#pragma unroll
    for (int r = 0; r < 8; ++r) {       // C: VGPR r -> M = r + 8*g, N = lane&15
      int o = o_base + r + 8 * gg;
      if (o < ODIM)
        out[(size_t)bb * (ODIM * SPAT) + (size_t)o * SPAT + ss] = acc[r] + bq[o];
    }
  }
}

// ---------------------------------------------------------------------------
// 2) depthwise 3x3 conv, pad 1, groups = 9000
// ---------------------------------------------------------------------------
__global__ __launch_bounds__(256) void dwconv(
    const float* __restrict__ in, const float* __restrict__ w,
    const float* __restrict__ bias, float* __restrict__ outp)
{
  int i = blockIdx.x * 256 + threadIdx.x;
  if (i >= BATCH * ODIM * SPAT) return;
  int s = i % SPAT; int o = (i / SPAT) % ODIM;
  int yy = s / 7, xx = s % 7;
  const float* base = in + (size_t)(i - s);
  const float* wf = w + o * 9;
  float a = bias[o];
#pragma unroll
  for (int dy = 0; dy < 3; ++dy) {
    int py = yy + dy - 1;
    if (py < 0 || py > 6) continue;
#pragma unroll
    for (int dx = 0; dx < 3; ++dx) {
      int px = xx + dx - 1;
      if (px < 0 || px > 6) continue;
      a += base[py * 7 + px] * wf[dy * 3 + dx];
    }
  }
  outp[i] = a;
}

// ---------------------------------------------------------------------------
// 3) L2 normalize q and k rows (first 6000 channels) in place over s=49
// ---------------------------------------------------------------------------
__global__ __launch_bounds__(256) void l2norm_qk(float* __restrict__ qkv)
{
  int i = blockIdx.x * 256 + threadIdx.x;
  if (i >= BATCH * 6000) return;
  int b = i / 6000, o = i % 6000;
  float* p = qkv + ((size_t)b * ODIM + o) * SPAT;
  float s = 0.f;
#pragma unroll 7
  for (int j = 0; j < SPAT; ++j) s += p[j] * p[j];
  float inv = 1.f / fmaxf(sqrtf(s), 1e-12f);
#pragma unroll 7
  for (int j = 0; j < SPAT; ++j) p[j] *= inv;
}

// ---------------------------------------------------------------------------
// 4) fused attention per (b,h, 16-row c-tile); k,v,q,P in LDS (~164 KB)
//    f32 WMMA 16x16x4 with scalar fallback; writes t[b,g,s,c]
// ---------------------------------------------------------------------------
#define KP 52
#define DP 304
#define VP 64

__global__ __launch_bounds__(256) void attention_fused(
    const float* __restrict__ qkv, const float* __restrict__ temp,
    float* __restrict__ tout)
{
  extern __shared__ float sm[];
  float* kls    = sm;                  // [304][52]
  float* vls    = kls + DP * KP;       // [300][64]
  float* qls    = vls + CH * VP;       // [16][52]
  float* pls    = qls + 16 * KP;       // [16][304]
  float* red    = pls + 16 * DP;       // [256]
  float* rowred = red + 256;           // [16]

  const int tid = threadIdx.x;
  const int bh = blockIdx.y;
  const int b = bh / HEADS, h = bh % HEADS;
  const int c0 = blockIdx.x * 16;
  const float* qp = qkv + ((size_t)b * ODIM + h * CH) * SPAT;
  const float* kp = qkv + ((size_t)b * ODIM + 3000 + h * CH) * SPAT;
  const float* vp = qkv + ((size_t)b * ODIM + 6000 + h * CH) * SPAT;
  const float tscale = temp[h];

  for (int i = tid; i < DP * KP; i += 256) {
    int d = i / KP, s = i % KP;
    kls[i] = (d < CH && s < SPAT) ? kp[d * SPAT + s] : 0.f;
  }
  for (int i = tid; i < CH * VP; i += 256) {
    int d = i / VP, s = i % VP;
    vls[i] = (s < SPAT) ? vp[d * SPAT + s] : 0.f;
  }
  for (int i = tid; i < 16 * KP; i += 256) {
    int r = i / KP, s = i % KP, c = c0 + r;
    qls[i] = (c < CH && s < SPAT) ? qp[c * SPAT + s] : 0.f;
  }
  __syncthreads();

  const int lane = tid & 31, wave = tid >> 5;
  const int mm = lane & 15, gg = lane >> 4;

#if USE_WMMA_F32
  for (int dt = wave; dt < 19; dt += 8) {
    v8f acc = {0.f,0.f,0.f,0.f,0.f,0.f,0.f,0.f};
    for (int ks = 0; ks < KP; ks += 4) {
      v2f av, bv;
      av[0] = qls[mm * KP + ks + 0 + 2 * gg];
      av[1] = qls[mm * KP + ks + 1 + 2 * gg];
      bv[0] = kls[(dt * 16 + mm) * KP + ks + 0 + 2 * gg];
      bv[1] = kls[(dt * 16 + mm) * KP + ks + 1 + 2 * gg];
      acc = __builtin_amdgcn_wmma_f32_16x16x4_f32(false, av, false, bv,
                                                  (short)0, acc, false, false);
    }
#pragma unroll
    for (int r = 0; r < 8; ++r)
      pls[(r + 8 * gg) * DP + dt * 16 + mm] = acc[r] * tscale;
  }
#else
  for (int i = tid; i < 16 * DP; i += 256) {
    int r = i / DP, d = i % DP;
    float a = 0.f;
    if (d < CH)
      for (int s = 0; s < SPAT; ++s) a += qls[r * KP + s] * kls[d * KP + s];
    pls[i] = a * tscale;
  }
#endif
  __syncthreads();

  { // softmax over d<300, 16 threads per row
    int r = tid >> 4, tc = tid & 15;
    float lmax = -3.0e38f;
    for (int d = tc; d < CH; d += 16) lmax = fmaxf(lmax, pls[r * DP + d]);
    red[tid] = lmax;
    __syncthreads();
    if (tc == 0) {
      float mv = red[r * 16];
      for (int i = 1; i < 16; ++i) mv = fmaxf(mv, red[r * 16 + i]);
      rowred[r] = mv;
    }
    __syncthreads();
    float mv = rowred[r], lsum = 0.f;
    for (int d = tc; d < CH; d += 16) {
      float e = expf(pls[r * DP + d] - mv);
      pls[r * DP + d] = e;
      lsum += e;
    }
    red[tid] = lsum;
    __syncthreads();
    if (tc == 0) {
      float sv = 0.f;
      for (int i = 0; i < 16; ++i) sv += red[r * 16 + i];
      rowred[r] = sv;
    }
    __syncthreads();
    float inv = 1.f / rowred[r];
    for (int d = tc; d < CH; d += 16) pls[r * DP + d] *= inv;
  }
  __syncthreads();

  float* trow = tout + (size_t)bh * SPAT * CH;
#if USE_WMMA_F32
  for (int st = wave; st < 4; st += 8) {
    v8f acc = {0.f,0.f,0.f,0.f,0.f,0.f,0.f,0.f};
    for (int kd = 0; kd < CH; kd += 4) {
      v2f av, bv;
      av[0] = pls[mm * DP + kd + 0 + 2 * gg];
      av[1] = pls[mm * DP + kd + 1 + 2 * gg];
      bv[0] = vls[(kd + 0 + 2 * gg) * VP + st * 16 + mm];
      bv[1] = vls[(kd + 1 + 2 * gg) * VP + st * 16 + mm];
      acc = __builtin_amdgcn_wmma_f32_16x16x4_f32(false, av, false, bv,
                                                  (short)0, acc, false, false);
    }
#pragma unroll
    for (int r = 0; r < 8; ++r) {
      int c = c0 + r + 8 * gg;
      int s = st * 16 + mm;
      if (c < CH && s < SPAT) trow[(size_t)s * CH + c] = acc[r];
    }
  }
#else
  for (int i = tid; i < 16 * SPAT; i += 256) {
    int r = i / SPAT, s = i % SPAT, c = c0 + r;
    if (c < CH) {
      float a = 0.f;
      for (int d = 0; d < CH; ++d) a += pls[r * DP + d] * vls[d * VP + s];
      trow[(size_t)s * CH + c] = a;
    }
  }
#endif
}

// ---------------------------------------------------------------------------
// 5) LayerNorm over c=300, one wave per row
// ---------------------------------------------------------------------------
__global__ __launch_bounds__(256) void ln_rows(
    const float* __restrict__ t, float* __restrict__ y,
    const float* __restrict__ gp, const float* __restrict__ bp, int nrows)
{
  int wid = threadIdx.x >> 5, lane = threadIdx.x & 31;
  int row = blockIdx.x * 8 + wid;
  if (row >= nrows) return;
  const float* tr = t + (size_t)row * CH;
  float s1 = 0.f, s2 = 0.f;
  for (int c = lane; c < CH; c += 32) { float v = tr[c]; s1 += v; s2 += v * v; }
  for (int off = 16; off; off >>= 1) {
    s1 += __shfl_xor(s1, off, 32);
    s2 += __shfl_xor(s2, off, 32);
  }
  float mv  = s1 * (1.f / CH);
  float var = s2 * (1.f / CH) - mv * mv;
  float inv = rsqrtf(var + 1e-5f);
  float* yr = y + (size_t)row * CH;
  for (int c = lane; c < CH; c += 32) yr[c] = (tr[c] - mv) * inv * gp[c] + bp[c];
}

// ---------------------------------------------------------------------------
// 6) token MLP 49->9->49 along s, residual add; one thread per (b,g,c)
// ---------------------------------------------------------------------------
__global__ __launch_bounds__(256) void token_mlp(
    const float* __restrict__ y, float* __restrict__ t,
    const float* __restrict__ wt1, const float* __restrict__ bt1,
    const float* __restrict__ wt2, const float* __restrict__ bt2)
{
  int i = blockIdx.x * blockDim.x + threadIdx.x;
  if (i >= NBGC) return;
  int c = i % CH; int bg = i / CH;
  size_t base = (size_t)bg * (SPAT * CH) + c;
  float yv[SPAT];
  for (int s = 0; s < SPAT; ++s) yv[s] = y[base + (size_t)s * CH];
  float hv[9];
#pragma unroll
  for (int j = 0; j < 9; ++j) {
    float a = bt1[j];
    for (int s = 0; s < SPAT; ++s) a += yv[s] * wt1[s * 9 + j];
    hv[j] = gelu_exact(a);
  }
  for (int s = 0; s < SPAT; ++s) {
    float a = bt2[s];
#pragma unroll
    for (int j = 0; j < 9; ++j) a += hv[j] * wt2[j * SPAT + s];
    t[base + (size_t)s * CH] += a;
  }
}

// ---------------------------------------------------------------------------
// 7) LN2 + channel MLP 300->60->300, residual add; one 64-thread block per row
// ---------------------------------------------------------------------------
__global__ __launch_bounds__(64) void ln_chan_mlp(
    float* __restrict__ t,
    const float* __restrict__ g2, const float* __restrict__ b2,
    const float* __restrict__ wc1, const float* __restrict__ bc1,
    const float* __restrict__ wc2, const float* __restrict__ bc2)
{
  __shared__ float yr[CH];
  __shared__ float hh[60];
  __shared__ float red[128];
  __shared__ float stats[2];
  int tid = threadIdx.x;
  float* tr = t + (size_t)blockIdx.x * CH;
  float s1 = 0.f, s2 = 0.f;
  for (int c = tid; c < CH; c += 64) { float v = tr[c]; s1 += v; s2 += v * v; }
  red[tid] = s1; red[64 + tid] = s2;
  __syncthreads();
  if (tid == 0) {
    float a = 0.f, q = 0.f;
    for (int k = 0; k < 64; ++k) { a += red[k]; q += red[64 + k]; }
    float mv = a * (1.f / CH);
    stats[0] = mv;
    stats[1] = rsqrtf(q * (1.f / CH) - mv * mv + 1e-5f);
  }
  __syncthreads();
  float mv = stats[0], inv = stats[1];
  for (int c = tid; c < CH; c += 64) yr[c] = (tr[c] - mv) * inv * g2[c] + b2[c];
  __syncthreads();
  if (tid < 60) {
    float a = bc1[tid];
    for (int c = 0; c < CH; ++c) a += yr[c] * wc1[c * 60 + tid];
    hh[tid] = gelu_exact(a);
  }
  __syncthreads();
  for (int c = tid; c < CH; c += 64) {
    float a = bc2[c];
#pragma unroll 10
    for (int k = 0; k < 60; ++k) a += hh[k] * wc2[k * CH + c];
    tr[c] += a;
  }
}

// ---------------------------------------------------------------------------
// 8) pooling
// ---------------------------------------------------------------------------
__global__ __launch_bounds__(256) void pool_avg(
    const float* __restrict__ t, float* __restrict__ out)
{
  int i = blockIdx.x * 256 + threadIdx.x;
  if (i >= NBGC) return;
  int c = i % CH, bg = i / CH;
  const float* p = t + (size_t)bg * (SPAT * CH) + c;
  float a = 0.f;
  for (int s = 0; s < SPAT; ++s) a += p[(size_t)s * CH];
  out[i] = a * (1.f / SPAT);
}

__global__ __launch_bounds__(256) void pool_max_sigmoid(
    const float* __restrict__ t, float* __restrict__ out)
{
  int i = blockIdx.x * 256 + threadIdx.x;
  if (i >= NBGC) return;
  int c = i % CH, bg = i / CH;
  const float* p = t + (size_t)bg * (SPAT * CH) + c;
  float m = -3.0e38f;
  for (int s = 0; s < SPAT; ++s) m = fmaxf(m, p[(size_t)s * CH]);
  float v = out[i] + m;
  out[i] = 1.f / (1.f + expf(-v));
}

// ---------------------------------------------------------------------------
extern "C" void kernel_launch(void* const* d_in, const int* in_sizes, int n_in,
                              void* d_out, int out_size, void* d_ws, size_t ws_size,
                              hipStream_t stream)
{
  const float* x    = (const float*)d_in[0];
  const float* temp = (const float*)d_in[1];
  const float* wq   = (const float*)d_in[2];
  const float* bq   = (const float*)d_in[3];
  const float* wdw  = (const float*)d_in[4];
  const float* bdw  = (const float*)d_in[5];
  const float* ln1g = (const float*)d_in[6];
  const float* ln1b = (const float*)d_in[7];
  const float* ln2g = (const float*)d_in[8];
  const float* ln2b = (const float*)d_in[9];
  const float* wt1  = (const float*)d_in[10];
  const float* bt1  = (const float*)d_in[11];
  const float* wt2  = (const float*)d_in[12];
  const float* bt2  = (const float*)d_in[13];
  const float* wc1  = (const float*)d_in[14];
  const float* bc1  = (const float*)d_in[15];
  const float* wc2  = (const float*)d_in[16];
  const float* bc2  = (const float*)d_in[17];
  float* out = (float*)d_out;
  float* ws  = (float*)d_ws;

  float*  qkv0 = ws;                       // 28,224,000 floats
  float*  qkv1 = ws + 28224000;            // 28,224,000 floats
  __bf16* wbf  = (__bf16*)qkv1;            // 27,721,728 bf16 (aliases qkv1)
  __bf16* xbf  = wbf + (size_t)OPAD * KPAD;//  9,433,088 bf16
  float*  t    = ws;                       //  9,408,000 floats (aliases qkv0)
  float*  y    = ws + 9408000;             //  9,408,000 floats

  // 0) stage bf16 operands (padded, x transposed to [n][k])
  conv_w_bf16<<<(OPAD * (KPAD / 8) + 255) / 256, 256, 0, stream>>>(wq, wbf);
  conv_x_bf16<<<(NTOT * (KPAD / 8) + 255) / 256, 256, 0, stream>>>(x, xbf);
  // 1) qkv 1x1 conv (bf16 WMMA GEMM, LDS-blocked, async copies)
  qkv_gemm<<<dim3(NTOT / 64, OPAD / 128), 256, 0, stream>>>(wbf, xbf, bq, qkv0);
  // 2) depthwise 3x3 (overwrites qkv1 region; wbf/xbf dead by now)
  dwconv<<<(BATCH * ODIM * SPAT) / 256, 256, 0, stream>>>(qkv0, wdw, bdw, qkv1);
  // 3) L2 normalize q,k
  l2norm_qk<<<(BATCH * 6000) / 256, 256, 0, stream>>>(qkv1);
  // 4) fused attention -> t (b,g,s,c)
  size_t shmem = (size_t)(DP * KP + CH * VP + 16 * KP + 16 * DP + 256 + 16) * sizeof(float);
  attention_fused<<<dim3(19, BATCH * HEADS), 256, shmem, stream>>>(qkv1, temp, t);
  // 5-7) mixer #1
  ln_rows<<<NROWS / 8, 256, 0, stream>>>(t, y, ln1g, ln1b, NROWS);
  token_mlp<<<NBGC / 256, 256, 0, stream>>>(y, t, wt1, bt1, wt2, bt2);
  ln_chan_mlp<<<NROWS, 64, 0, stream>>>(t, ln2g, ln2b, wc1, bc1, wc2, bc2);
  // 8) avg pool into out (pre-sigmoid partial)
  pool_avg<<<NBGC / 256, 256, 0, stream>>>(t, out);
  // 9-11) mixer #2
  ln_rows<<<NROWS / 8, 256, 0, stream>>>(t, y, ln1g, ln1b, NROWS);
  token_mlp<<<NBGC / 256, 256, 0, stream>>>(y, t, wt1, bt1, wt2, bt2);
  ln_chan_mlp<<<NROWS, 64, 0, stream>>>(t, ln2g, ln2b, wc1, bc1, wc2, bc2);
  // 12) max pool + sigmoid -> final output
  pool_max_sigmoid<<<NBGC / 256, 256, 0, stream>>>(t, out);

  (void)in_sizes; (void)n_in; (void)out_size; (void)ws_size;
}
